// PathConvLR_14164802142829
// MI455X (gfx1250) — matrix operand order repaired and checked by reading, not compiled, not run on hardware
//
#include <hip/hip_runtime.h>

// ============================================================================
// PathConvLR inference for MI455X (gfx1250, wave32, WMMA).
//
// Roofline: ~0.68 TFLOP of pointwise-conv GEMM work vs ~GBs of activation
// traffic (23.3 TB/s HBM). The 1x1 convs + head run as bf16 WMMA GEMMs
// (v_wmma_f32_16x16x32_bf16, fp32 accumulate):
//   - X (B-operand) staged transposed into LDS with f32->bf16 convert,
//     branch-free fast path + guarded tail, 2x ds_store_b128 per thread.
//   - W (A-operand) tile staged into LDS via CDNA5 async DMA
//     (global_load_async_to_lds_b128 + s_wait_asynccnt) on full tiles,
//     guarded VGPR fallback on edge tiles.
// Depthwise convs, LNs, ECA, pooling are bandwidth-bound VALU kernels with
// coalesced access.
//
// Input leaf order assumes jax.tree_util.tree_leaves on the setup_inputs()
// dict (dict keys sorted alphabetically at every nesting level, lists in
// order) => 277 leaves; index map in kernel_launch.
// ============================================================================

typedef __bf16 bf16_t;
typedef bf16_t v16bf __attribute__((ext_vector_type(16)));
typedef bf16_t v8bf  __attribute__((ext_vector_type(8)));
typedef float  v8f   __attribute__((ext_vector_type(8)));

#define NB 256          // batch
#define NPATH 4

__device__ __forceinline__ float gelu_exact(float v) {
  return 0.5f * v * (1.0f + erff(v * 0.70710678118654752f));
}
__device__ __forceinline__ float sigmoidf(float v) {
  return 1.0f / (1.0f + expf(-v));
}

// ---------------------------------------------------------------------------
// Path gather: out[b, p*3+c, l] = x[b, c, paths[p,l,0]*32 + paths[p,l,1]]
// ---------------------------------------------------------------------------
__global__ void k_gather(const float* __restrict__ x, const int* __restrict__ paths,
                         float* __restrict__ out) {
  int l  = blockIdx.x * blockDim.x + threadIdx.x;   // 0..1023
  int pc = blockIdx.y;                              // 0..11
  int b  = blockIdx.z;
  int p = pc / 3, c = pc % 3;
  int r  = paths[(p * 1024 + l) * 2 + 0];
  int cc = paths[(p * 1024 + l) * 2 + 1];
  out[((size_t)b * 12 + pc) * 1024 + l] = x[(((size_t)b * 3 + c) * 32 + r) * 32 + cc];
}

// ---------------------------------------------------------------------------
// WMMA GEMM:  Out[b, o, l] = sum_k Wm[o, k] * X[b, k, l]  (+ bias / epilogues)
//   mode 0: Out = acc + bias
//   mode 1: Out = (acc + bias) * scale[o] + resid[b, o, l]
//   mode 2: Out[l * O + o] = acc + bias            (transposed store; head)
// Block tile 64(O) x 128(L), 8 waves: 4 in O, 2 in L, 4 accumulators/wave.
// Requires L % 128 == 0 (true at all call sites: 1024/512/256/128/256).
// ---------------------------------------------------------------------------
__global__ __launch_bounds__(256) void k_gemm(
    const float* __restrict__ X, const float* __restrict__ Wm,
    const float* __restrict__ bias, float* __restrict__ Out,
    const float* __restrict__ scale, const float* __restrict__ resid,
    int K, int O, int L, int Kp, int mode)
{
  // X tile transposed: Xs[n][k], bf16. Row stride 40 halves = 80B (16B mult).
  __shared__ __align__(16) bf16_t Xs[128][40];
  // W tile: Wls[o][k], f32. Row stride 36 floats = 144B (16B mult).
  __shared__ __align__(16) float Wls[64][36];

  const int tid  = threadIdx.x;
  const int lane = tid & 31;
  const int wave = tid >> 5;
  const int wy   = wave >> 1;      // 0..3 : O sub-tile
  const int wx   = wave & 1;       // 0..1 : L sub-tile
  const int b    = blockIdx.z;
  const int o0   = blockIdx.y * 64;
  const int l0   = blockIdx.x * 128;

  v8f acc[4];
#pragma unroll
  for (int t = 0; t < 4; ++t)
#pragma unroll
    for (int i = 0; i < 8; ++i) acc[t][i] = 0.0f;

  // Staging thread mappings.
  const int sn  = tid & 127;          // X: n within tile
  const int skh = (tid >> 7) * 16;    // X: k half (0 or 16)
  const int wr  = tid >> 2;           // W: row 0..63
  const int wq  = tid & 3;            // W: col quad (8 floats)

  const float* xcol = X + (size_t)b * K * L + l0 + sn;

  for (int c0 = 0; c0 < Kp; c0 += 32) {
    __syncthreads();

    // ---- Stage X tile (transpose + f32->bf16), coalesced over n ----
    {
      const float* xp = xcol + (size_t)(c0 + skh) * L;
      v8bf lo, hi;
      if (c0 + 32 <= K) {               // fast path: no guards
#pragma unroll
        for (int kk = 0; kk < 8; ++kk) lo[kk] = (bf16_t)xp[(size_t)kk * L];
#pragma unroll
        for (int kk = 0; kk < 8; ++kk) hi[kk] = (bf16_t)xp[(size_t)(kk + 8) * L];
        if (c0 + 32 < Kp)
          __builtin_prefetch(xp + (size_t)32 * L, 0, 1);
      } else {                          // guarded tail (last K-step only)
#pragma unroll
        for (int kk = 0; kk < 8; ++kk)
          lo[kk] = (c0 + skh + kk < K) ? (bf16_t)xp[(size_t)kk * L] : (bf16_t)0.0f;
#pragma unroll
        for (int kk = 0; kk < 8; ++kk)
          hi[kk] = (c0 + skh + kk + 8 < K) ? (bf16_t)xp[(size_t)(kk + 8) * L] : (bf16_t)0.0f;
      }
      *(v8bf*)&Xs[sn][skh]     = lo;
      *(v8bf*)&Xs[sn][skh + 8] = hi;
    }

    // ---- Stage W tile ----
    if (o0 + 64 <= O && c0 + 32 <= K) {
      // Full tile: CDNA5 async DMA global -> LDS (2 x 16B per thread),
      // contiguous-to-contiguous, no VGPR round trip. INST_OFFSET applies to
      // both the global and LDS addresses per the ISA, so offset:16 lands the
      // second beat in the right place.
      unsigned long long ga =
          (unsigned long long)(uintptr_t)(Wm + (size_t)(o0 + wr) * K + c0 + wq * 8);
      unsigned int la = (unsigned int)(uintptr_t)&Wls[wr][wq * 8];
      asm volatile("global_load_async_to_lds_b128 %0, %1, off"
                   :: "v"(la), "v"(ga) : "memory");
      asm volatile("global_load_async_to_lds_b128 %0, %1, off offset:16"
                   :: "v"(la), "v"(ga) : "memory");
      asm volatile("s_wait_asynccnt 0x0" ::: "memory");
    } else {
      // Edge tile: guarded VGPR path writing zeros out of range.
      int orow2 = o0 + wr;
#pragma unroll
      for (int j = 0; j < 8; ++j) {
        int c = c0 + wq * 8 + j;
        Wls[wr][wq * 8 + j] =
            (orow2 < O && c < K) ? Wm[(size_t)orow2 * K + c] : 0.0f;
      }
    }
    __syncthreads();

    // ---- A fragment (16x32 bf16) from LDS W tile, per-ISA lane layout ----
    const int khalf = (lane < 16) ? 0 : 8;
    const float* warow = &Wls[wy * 16 + (lane & 15)][0];
    v16bf a;
#pragma unroll
    for (int j = 0; j < 8; ++j) a[j]     = (bf16_t)warow[khalf + j];
#pragma unroll
    for (int j = 0; j < 8; ++j) a[j + 8] = (bf16_t)warow[16 + khalf + j];

    // ---- B fragments first (overlap ds_loads), then the 4 WMMAs ----
    v16bf bb[4];
#pragma unroll
    for (int t = 0; t < 4; ++t) {
      const bf16_t* bp = &Xs[wx * 64 + t * 16 + (lane & 15)][(lane < 16) ? 0 : 8];
      v8bf u0 = *(const v8bf*)bp;
      v8bf u1 = *(const v8bf*)(bp + 16);
#pragma unroll
      for (int i = 0; i < 8; ++i) { bb[t][i] = u0[i]; bb[t][i + 8] = u1[i]; }
    }
#pragma unroll
    for (int t = 0; t < 4; ++t)
      acc[t] = __builtin_amdgcn_wmma_f32_16x16x32_bf16(
          false, a, false, bb[t], (short)0, acc[t], false, false);
  }

  // Epilogue. C/D layout: VGPR r -> row r (lanes 0-15) / r+8 (lanes 16-31).
#pragma unroll
  for (int t = 0; t < 4; ++t) {
    int l = l0 + wx * 64 + t * 16 + (lane & 15);
#pragma unroll
    for (int r = 0; r < 8; ++r) {
      int o = o0 + wy * 16 + r + ((lane < 16) ? 0 : 8);
      if (o < O) {
        float v = acc[t][r];
        if (bias) v += bias[o];
        if (mode == 1)
          v = v * scale[o] + resid[(size_t)b * O * L + (size_t)o * L + l];
        if (mode == 2)
          Out[(size_t)l * O + o] = v;
        else
          Out[(size_t)b * O * L + (size_t)o * L + l] = v;
      }
    }
  }
}

// ---------------------------------------------------------------------------
// Depthwise / grouped-expand conv1d. g = co / expand (expand = Cout/Cin).
// Optional per-channel pre-scale of the input: in *= (1 + scaleIn[b, g])
// (hoisted out of the tap loop since it is constant per channel).
// ---------------------------------------------------------------------------
__global__ void k_dwconv(const float* __restrict__ in, const float* __restrict__ w,
                         const float* __restrict__ bias, const float* __restrict__ scaleIn,
                         float* __restrict__ out,
                         int Cin, int Cout, int Lin, int Lout,
                         int ksz, int pad, int stride, int expand)
{
  int l  = blockIdx.x * blockDim.x + threadIdx.x;
  int co = blockIdx.y;
  int b  = blockIdx.z;
  if (l >= Lout) return;
  int g = co / expand;
  const float* ip = in + ((size_t)b * Cin + g) * Lin;
  float s = 0.f;
  for (int k = 0; k < ksz; ++k) {
    int li = l * stride + k - pad;
    if (li >= 0 && li < Lin) s += w[co * ksz + k] * ip[li];
  }
  if (scaleIn) s *= (1.0f + scaleIn[(size_t)b * Cin + g]);
  if (bias) s += bias[co];
  out[((size_t)b * Cout + co) * Lout + l] = s;
}

// ---------------------------------------------------------------------------
// Channel-first LayerNorm (reduce over C per (b,l)), in place.
// Optional fused exact GELU and fused positional add (addpos[c*L+l]).
// Block = 32 l-columns x 8 c-parallel threads; coalesced over l.
// ---------------------------------------------------------------------------
__global__ __launch_bounds__(256) void k_ln_cf(
    float* __restrict__ x, const float* __restrict__ w, const float* __restrict__ bvec,
    int C, int L, const float* __restrict__ addpos, int dogelu, float eps)
{
  __shared__ float rs[8][33], rq[8][33];
  __shared__ float mS[32], iS[32];
  int tx = threadIdx.x & 31, ty = threadIdx.x >> 5;
  int l = blockIdx.x * 32 + tx;
  int b = blockIdx.z;
  float* base = x + (size_t)b * C * L + l;
  float s = 0.f, q = 0.f;
  for (int c = ty; c < C; c += 8) {
    float v = base[(size_t)c * L];
    s += v; q += v * v;
  }
  rs[ty][tx] = s; rq[ty][tx] = q;
  __syncthreads();
  if (ty == 0) {
    float ss = 0.f, qq = 0.f;
#pragma unroll
    for (int i = 0; i < 8; ++i) { ss += rs[i][tx]; qq += rq[i][tx]; }
    float mean = ss / C;
    float var  = qq / C - mean * mean;
    mS[tx] = mean;
    iS[tx] = rsqrtf(var + eps);
  }
  __syncthreads();
  float mean = mS[tx], inv = iS[tx];
  for (int c = ty; c < C; c += 8) {
    float v = base[(size_t)c * L];
    float y = w[c] * (v - mean) * inv + bvec[c];
    if (dogelu) y = gelu_exact(y);
    if (addpos) y += addpos[(size_t)c * L + l];
    base[(size_t)c * L] = y;
  }
}

// ---------------------------------------------------------------------------
// Mean over L: y[b*C + c] = mean_l x[b, c, l]. One wave32 per (b,c) row.
// ---------------------------------------------------------------------------
__global__ void k_poolL(const float* __restrict__ x, float* __restrict__ y,
                        int L, int total)
{
  int lane = threadIdx.x & 31;
  int wv   = threadIdx.x >> 5;
  int idx  = blockIdx.x * 8 + wv;
  if (idx >= total) return;
  const float* p = x + (size_t)idx * L;
  float s = 0.f;
  for (int i = lane; i < L; i += 32) s += p[i];
#pragma unroll
  for (int off = 16; off > 0; off >>= 1) s += __shfl_down(s, off, 32);
  if (lane == 0) y[idx] = s / (float)L;
}

// ---------------------------------------------------------------------------
// EnhancedPathAwareECA (tiny, one block per batch element).
// y: pooled means [B, C]. Produces sOut[b, c] = attn * cross-path-weight.
// (Caller applies x*(1+s) -- folded into the next depthwise conv.)
// ---------------------------------------------------------------------------
__global__ __launch_bounds__(128) void k_eca(
    const float* __restrict__ y, const float* __restrict__ pw1, const float* __restrict__ pw2,
    const float* __restrict__ comb_w, const float* __restrict__ comb_b,
    const float* __restrict__ ln_w, const float* __restrict__ ln_b,
    const float* __restrict__ w1, const float* __restrict__ b1,
    const float* __restrict__ w2, const float* __restrict__ b2,
    float* __restrict__ sOut, int C)
{
  __shared__ float attnS[480];
  __shared__ float cpS[4];
  __shared__ float cwS[4];
  int b = blockIdx.x, tid = threadIdx.x;
  int dpp = C >> 2;
  if (tid < 4) cpS[tid] = 0.f;
  __syncthreads();
  const float* yb = y + (size_t)b * C;
  for (int c = tid; c < C; c += 128) {
    int p = c / dpp, j = c - p * dpp;
    float y1 = 0.f, y2 = 0.f;
#pragma unroll
    for (int t = 0; t < 3; ++t) {
      int jj = j + t - 1;
      if (jj >= 0 && jj < dpp) y1 += pw1[p * 3 + t] * yb[p * dpp + jj];
    }
#pragma unroll
    for (int t = 0; t < 5; ++t) {
      int jj = j + t - 2;
      if (jj >= 0 && jj < dpp) y2 += pw2[p * 5 + t] * yb[p * dpp + jj];
    }
    float a = sigmoidf(comb_w[0] * y1 + comb_w[1] * y2 + comb_b[0]);
    attnS[c] = a;
    atomicAdd(&cpS[p], a);
  }
  __syncthreads();
  if (tid == 0) {
    float cp[4];
#pragma unroll
    for (int p2 = 0; p2 < 4; ++p2) cp[p2] = cpS[p2] / (float)dpp;
    float u = 0.25f * (cp[0] + cp[1] + cp[2] + cp[3]);
    float var = 0.f;
#pragma unroll
    for (int p2 = 0; p2 < 4; ++p2) { float d = cp[p2] - u; var += d * d; }
    var *= 0.25f;
    float inv = rsqrtf(var + 1e-5f);
    float h[4], hh[8];
#pragma unroll
    for (int p2 = 0; p2 < 4; ++p2) h[p2] = ln_w[p2] * (cp[p2] - u) * inv + ln_b[p2];
#pragma unroll
    for (int i = 0; i < 8; ++i) {
      float t = b1[i];
#pragma unroll
      for (int p2 = 0; p2 < 4; ++p2) t += w1[i * 4 + p2] * h[p2];
      hh[i] = gelu_exact(t);
    }
#pragma unroll
    for (int p2 = 0; p2 < 4; ++p2) {
      float t = b2[p2];
#pragma unroll
      for (int i = 0; i < 8; ++i) t += w2[p2 * 8 + i] * hh[i];
      cwS[p2] = sigmoidf(t);
    }
  }
  __syncthreads();
  for (int c = tid; c < C; c += 128)
    sOut[(size_t)b * C + c] = attnS[c] * cwS[c / dpp];
}

// ---------------------------------------------------------------------------
// Last-dim LayerNorm of pooled features + transposed store for the head GEMM:
// outT[c * B + b] = w[c] * (f[b,c] - u) * inv + b[c].
// ---------------------------------------------------------------------------
__global__ __launch_bounds__(256) void k_lnlast_t(
    const float* __restrict__ f, const float* __restrict__ w, const float* __restrict__ bvec,
    float* __restrict__ outT, int Bn, int C, float eps)
{
  __shared__ float rs[256], rq[256];
  __shared__ float stats[2];
  int b = blockIdx.x, tid = threadIdx.x;
  const float* p = f + (size_t)b * C;
  float s = 0.f, q = 0.f;
  for (int c = tid; c < C; c += 256) { float v = p[c]; s += v; q += v * v; }
  rs[tid] = s; rq[tid] = q;
  __syncthreads();
  for (int off = 128; off > 0; off >>= 1) {
    if (tid < off) { rs[tid] += rs[tid + off]; rq[tid] += rq[tid + off]; }
    __syncthreads();
  }
  if (tid == 0) {
    float mean = rs[0] / C;
    float var  = rq[0] / C - mean * mean;
    stats[0] = mean; stats[1] = rsqrtf(var + eps);
  }
  __syncthreads();
  float mean = stats[0], inv = stats[1];
  for (int c = tid; c < C; c += 256)
    outT[(size_t)c * Bn + b] = w[c] * (p[c] - mean) * inv + bvec[c];
}

// ===========================================================================
// Host orchestration
// ===========================================================================
static inline void launch_gemm(const float* X, const float* Wm, const float* bias,
                               float* Out, const float* scale, const float* resid,
                               int K, int O, int L, int Bn, int mode, hipStream_t st) {
  int Kp = (K + 31) & ~31;
  dim3 g(L / 128, (O + 63) / 64, Bn);
  k_gemm<<<g, 256, 0, st>>>(X, Wm, bias, Out, scale, resid, K, O, L, Kp, mode);
}

extern "C" void kernel_launch(void* const* d_in, const int* in_sizes, int n_in,
                              void* d_out, int out_size, void* d_ws, size_t ws_size,
                              hipStream_t stream) {
  (void)in_sizes; (void)n_in; (void)out_size; (void)ws_size;
  auto F = [&](int i) { return (const float*)d_in[i]; };

  // ---- Leaf index map (jax tree_leaves order; dict keys sorted) ----
  // params.down[i]:   i*4 + {0:b, 1:ln_b, 2:ln_w, 3:w}          i=0..2
  // params.head:      12:b, 13:w
  // params.norm:      14:b, 15:w
  // params.pos:       16
  // params.stages[s][j]: base = 17 + (s*3+j)*21, offsets:
  //   0 comb_b, 1 comb_w, 2 mlp_b1, 3 mlp_b2, 4 mlp_ln_b, 5 mlp_ln_w,
  //   6 mlp_w1, 7 mlp_w2, 8 pw1, 9 pw2, 10 b1, 11 b2, 12 dw_b, 13 dw_ln_b,
  //   14 dw_ln_w, 15 dw_w, 16 ln2_b, 17 ln2_w, 18 scaling, 19 w1, 20 w2
  // params.stem:      269:b1, 270:b2, 271:ln_b, 272:ln_w, 273:w1, 274:w2
  // paths: 275 (int32), x: 276
  const int* paths  = (const int*)d_in[275];
  const float* x_in = F(276);

  // ---- Workspace carve-up ----
  char* ws = (char*)d_ws;
  const size_t ACT = (size_t)NB * 61440 * sizeof(float);  // 63 MB per activation
  float* bufX   = (float*)(ws);
  float* bufT   = (float*)(ws + ACT);
  float* bufE   = (float*)(ws + 2 * ACT);                 // 4*ACT (expand)
  float* bufPool= (float*)(ws + 6 * ACT);                 // [B, 480]
  float* bufS   = bufPool + (size_t)NB * 480;             // [B, 480]
  float* featT  = bufS    + (size_t)NB * 480;             // [480, B]

  const int DIMS_[4] = {60, 120, 240, 480};
  const int LS[4]    = {1024, 512, 256, 128};

  // ---- Stem ----
  k_gather<<<dim3(1024 / 256, 12, NB), 256, 0, stream>>>(x_in, paths, bufT);
  // 1x1 conv 12 -> 60 (WMMA, K padded 12->32)
  launch_gemm(bufT, F(273), F(269), bufX, nullptr, nullptr, 12, 60, 1024, NB, 0, stream);
  // depthwise k=11 pad=5
  k_dwconv<<<dim3(1024 / 256, 60, NB), 256, 0, stream>>>(
      bufX, F(274), F(270), nullptr, bufT, 60, 60, 1024, 1024, 11, 5, 1, 1);
  // LN (channel-first) + positional add, in place
  k_ln_cf<<<dim3(1024 / 32, 1, NB), 256, 0, stream>>>(
      bufT, F(272), F(271), 60, 1024, F(16), 0, 1e-6f);
  { float* t = bufX; bufX = bufT; bufT = t; }

  // ---- Stages ----
  for (int s = 0; s < 4; ++s) {
    const int d = DIMS_[s], L = LS[s];
    if (s > 0) {
      const int dp = DIMS_[s - 1], Lp = LS[s - 1];
      const int db = (s - 1) * 4;
      // grouped stride-2 expand conv (groups = dp, out = 2*dp), k=9 pad=4
      k_dwconv<<<dim3((L + 255) / 256, d, NB), 256, 0, stream>>>(
          bufX, F(db + 3), F(db + 0), nullptr, bufT, dp, d, Lp, L, 9, 4, 2, 2);
      k_ln_cf<<<dim3(L / 32, 1, NB), 256, 0, stream>>>(
          bufT, F(db + 2), F(db + 1), d, L, nullptr, 0, 1e-6f);
      { float* t = bufX; bufX = bufT; bufT = t; }
    }
    for (int j = 0; j < 3; ++j) {
      const int bb = 17 + (s * 3 + j) * 21;
      // ECA: pool over L, then per-batch attention -> per-channel scale s
      k_poolL<<<(NB * d + 7) / 8, 256, 0, stream>>>(bufX, bufPool, L, NB * d);
      k_eca<<<NB, 128, 0, stream>>>(bufPool, F(bb + 8), F(bb + 9),
                                    F(bb + 1), F(bb + 0), F(bb + 5), F(bb + 4),
                                    F(bb + 6), F(bb + 2), F(bb + 7), F(bb + 3),
                                    bufS, d);
      // depthwise k=9 on x*(1+s)  (scale folded in)
      k_dwconv<<<dim3((L + 255) / 256, d, NB), 256, 0, stream>>>(
          bufX, F(bb + 15), F(bb + 12), bufS, bufT, d, d, L, L, 9, 4, 1, 1);
      k_ln_cf<<<dim3(L / 32, 1, NB), 256, 0, stream>>>(
          bufT, F(bb + 14), F(bb + 13), d, L, nullptr, 0, 1e-6f);
      // 1x1 expand d -> 4d (WMMA)
      launch_gemm(bufT, F(bb + 19), F(bb + 10), bufE, nullptr, nullptr,
                  d, 4 * d, L, NB, 0, stream);
      // LN + exact GELU in place on expanded tensor
      k_ln_cf<<<dim3(L / 32, 1, NB), 256, 0, stream>>>(
          bufE, F(bb + 17), F(bb + 16), 4 * d, L, nullptr, 1, 1e-6f);
      // 1x1 project 4d -> d (WMMA) with scaling + residual(pre-ECA x)
      launch_gemm(bufE, F(bb + 20), F(bb + 11), bufT, F(bb + 18), bufX,
                  4 * d, d, L, NB, 1, stream);
      { float* t = bufX; bufX = bufT; bufT = t; }
    }
  }

  // ---- Head ----
  k_poolL<<<(NB * 480 + 7) / 8, 256, 0, stream>>>(bufX, bufPool, 128, NB * 480);
  k_lnlast_t<<<NB, 256, 0, stream>>>(bufPool, F(15), F(14), featT, NB, 480, 1e-6f);
  // Head GEMM: A = head.w [1000,480], X = featT [480, 256] viewed as [1,K,L];
  // mode 2 stores transposed directly into d_out [256, 1000].
  launch_gemm(featT, F(13), F(12), (float*)d_out, nullptr, nullptr,
              480, 1000, 256, 1, 2, stream);
}
